// DiscreteFlow_76656576299251
// MI455X (gfx1250) — compile-verified, matrix operand order checked by reading, not run on hardware
//
#include <hip/hip_runtime.h>
#include <hip/hip_bf16.h>

// ---------- sizes ----------
#define DD 16
#define KK 64
#define TT 1024
#define HH 1024
#define BB 32768
#define ROWS 64          // samples per block
#define LSTRIDE 1032     // padded bf16 row stride in LDS (1024 + 8)
#define XS_INTS (ROWS * DD)              // 1024 ints = 4KB x-tile in LDS

typedef __attribute__((ext_vector_type(16))) __bf16 v16bf;
typedef __attribute__((ext_vector_type(8)))  __bf16 v8bf;
typedef __attribute__((ext_vector_type(8)))  float  v8f;

// =====================================================================
// Prep: build masked, transposed bf16 weights in workspace.
//   W2t[n][k] = bf16( W2[k][n] * (n%15 >= k%15) )        (M2 mask)
//   W3t[t][k] = bf16( W3[k][t] * (t>>6  >  k%15) )       (M3 mask)
// =====================================================================
__global__ void made_prep_weights(const float* __restrict__ W2,
                                  const float* __restrict__ W3,
                                  __bf16* __restrict__ W2t,
                                  __bf16* __restrict__ W3t) {
  int idx = blockIdx.x * blockDim.x + threadIdx.x;  // 0 .. 1M-1
  int n = idx >> 10;
  int k = idx & 1023;
  if (blockIdx.y == 0) {
    float m = ((n % 15) >= (k % 15)) ? 1.0f : 0.0f;
    W2t[idx] = (__bf16)(W2[k * HH + n] * m);
  } else {
    float m = ((n >> 6) > (k % 15)) ? 1.0f : 0.0f;
    W3t[idx] = (__bf16)(W3[k * HH + n] * m);
  }
}

// Load one 16x32 bf16 WMMA fragment (A-style layout) from row-major data.
// base points at element (row, kk); koff = (lane>>4)*8.
// Per-lane: VGPRs 0-3 hold K[koff..koff+7], VGPRs 4-7 hold K[koff+16..koff+23].
__device__ __forceinline__ v16bf load_frag(const __bf16* base, int koff) {
  v8bf lo = *(const v8bf*)(base + koff);
  v8bf hi = *(const v8bf*)(base + koff + 16);
  v16bf r;
#pragma unroll
  for (int i = 0; i < 8; ++i) { r[i] = lo[i]; r[i + 8] = hi[i]; }
  return r;
}

// =====================================================================
// Fused MADE kernel: async-stage x tile -> gather-layer1 -> WMMA layer2
// -> WMMA layer3 + fused per-distribution log-softmax reduction.
// 512 threads = 16 waves; wave w owns N-tiles [4w,4w+4) == distribution
// d=w in stage 3. 64 samples per block live in LDS as bf16 activations.
// LDS layout (dynamic segment, byte offsets):
//   [0)        xs      : 64x16 int32 x-tile (4096 B)
//   [4096)     h1      : 64x1032 bf16
//   [+132096)  h2      : 64x1032 bf16
//   [+132096)  outAcc  : 64 float
// =====================================================================
__global__ void __launch_bounds__(512, 1)
made_fused(const int* __restrict__ x, const float* __restrict__ W1,
           const float* __restrict__ b1, const float* __restrict__ b2,
           const float* __restrict__ b3, const __bf16* __restrict__ W2t,
           const __bf16* __restrict__ W3t, float* __restrict__ out) {
  extern __shared__ char smem_raw[];
  int*    xs = (int*)smem_raw;                          // offset 0
  __bf16* h1 = (__bf16*)(smem_raw + sizeof(int) * XS_INTS);
  __bf16* h2 = h1 + ROWS * LSTRIDE;
  float* outAcc = (float*)(h2 + ROWS * LSTRIDE);

  const int tid  = threadIdx.x;
  const int wave = tid >> 5;
  const int lane = tid & 31;
  const int hi   = lane >> 4;        // half-wave select
  const int ln   = lane & 15;        // M (A) / N (B,C,D) index within tile
  const int koff = hi * 8;           // per-lane K offset for 16-bit frags
  const long rowBase = (long)blockIdx.x * ROWS;
  const v8f vzero = {};

  // ---- async-stage the 64x16 int x-tile into LDS (ASYNCcnt path) ----
  // 512 threads x 8 bytes = 4096 B = the whole tile, one b64 per lane.
  {
    unsigned ldsOff = (unsigned)tid * 8u;     // xs lives at LDS offset 0
    unsigned goff   = (unsigned)tid * 8u;     // byte offset into x tile
    const int* xblk = x + rowBase * DD;       // uniform -> SGPR pair
    asm volatile("global_load_async_to_lds_b64 %0, %1, %2"
                 :: "v"(ldsOff), "v"(goff), "s"(xblk)
                 : "memory");
    asm volatile("s_wait_asynccnt 0x0" ::: "memory");
  }

  if (tid < ROWS) outAcc[tid] = 0.0f;
  __syncthreads();   // xs visible to all waves

  // ---------------- stage 1: one-hot gather layer ----------------
  // h1[r][c] = relu(b1[c] + sum_{d<=c%15} W1[(d*64 + x[r][d])][c])
#pragma unroll
  for (int i = 0; i < 4; ++i) {
    int r = wave * 4 + i;
    int xr[15];
#pragma unroll
    for (int d = 0; d < 15; ++d) xr[d] = xs[r * DD + d];
    for (int cc = 0; cc < 32; ++cc) {
      int c = lane + cc * 32;        // coalesced across the wave
      int hd = c % 15;
      float acc = b1[c];
      for (int d = 0; d <= hd; ++d)
        acc += W1[(d * KK + xr[d]) * HH + c];
      h1[r * LSTRIDE + c] = (__bf16)fmaxf(acc, 0.0f);
    }
  }
  __syncthreads();

  // ---------------- stage 2: h2 = relu(h1 @ W2t^T + b2) ----------------
  {
    const int ntBase = wave * 4;     // 4 N-tiles per wave (64 cols)
    v8f c[4][4];
#pragma unroll
    for (int mt = 0; mt < 4; ++mt)
#pragma unroll
      for (int nt = 0; nt < 4; ++nt) c[mt][nt] = vzero;

    for (int kk = 0; kk < HH; kk += 32) {
      v16bf afr[4];
#pragma unroll
      for (int mt = 0; mt < 4; ++mt)
        afr[mt] = load_frag(&h1[(mt * 16 + ln) * LSTRIDE + kk], koff);
#pragma unroll
      for (int nt = 0; nt < 4; ++nt) {
        v16bf bfr = load_frag(&W2t[(size_t)((ntBase + nt) * 16 + ln) * HH + kk], koff);
#pragma unroll
        for (int mt = 0; mt < 4; ++mt)
          c[mt][nt] = __builtin_amdgcn_wmma_f32_16x16x32_bf16(
              false, afr[mt], false, bfr, (short)0, c[mt][nt], false, false);
      }
    }
#pragma unroll
    for (int nt = 0; nt < 4; ++nt) {
      int col = (ntBase + nt) * 16 + ln;
      float bias = b2[col];
#pragma unroll
      for (int mt = 0; mt < 4; ++mt)
#pragma unroll
        for (int v = 0; v < 8; ++v) {
          int row = mt * 16 + v + hi * 8;   // C/D layout: VGPR v -> M = v + 8*hi
          h2[row * LSTRIDE + col] = (__bf16)fmaxf(c[mt][nt][v] + bias, 0.0f);
        }
    }
  }
  __syncthreads();

  // ------- stage 3: logits = h2 @ W3t^T + b3, fused log-softmax -------
  {
    const int dIdx = wave;            // this wave's distribution
    const int colBase = dIdx * 64;    // its 64 logit columns (4 N-tiles)
    v8f c[4][4];
#pragma unroll
    for (int mt = 0; mt < 4; ++mt)
#pragma unroll
      for (int nt = 0; nt < 4; ++nt) c[mt][nt] = vzero;

    for (int kk = 0; kk < HH; kk += 32) {
      v16bf afr[4];
#pragma unroll
      for (int mt = 0; mt < 4; ++mt)
        afr[mt] = load_frag(&h2[(mt * 16 + ln) * LSTRIDE + kk], koff);
#pragma unroll
      for (int nt = 0; nt < 4; ++nt) {
        v16bf bfr = load_frag(&W3t[(size_t)(colBase + nt * 16 + ln) * HH + kk], koff);
#pragma unroll
        for (int mt = 0; mt < 4; ++mt)
          c[mt][nt] = __builtin_amdgcn_wmma_f32_16x16x32_bf16(
              false, afr[mt], false, bfr, (short)0, c[mt][nt], false, false);
      }
    }

    // per-row: out_partial = logits[k=x] - log(sum_k exp(logits[k]))
    float bias[4];
#pragma unroll
    for (int nt = 0; nt < 4; ++nt) bias[nt] = b3[colBase + nt * 16 + ln];

#pragma unroll
    for (int mt = 0; mt < 4; ++mt) {
#pragma unroll
      for (int v = 0; v < 8; ++v) {
        int row = mt * 16 + v + hi * 8;
        int kx = xs[row * DD + dIdx];   // observed index in [0,64)
        float esum = 0.0f, sel = 0.0f;
#pragma unroll
        for (int nt = 0; nt < 4; ++nt) {
          float lg = c[mt][nt][v] + bias[nt];
          esum += __expf(lg);
          if ((nt == (kx >> 4)) && (ln == (kx & 15))) sel = lg;
        }
        // 16-lane reduction; xor masks 1,2,4,8 stay within the half-wave,
        // so lanes 0-15 (row v) and 16-31 (row v+8) reduce independently.
#pragma unroll
        for (int off = 1; off < 16; off <<= 1) {
          esum += __shfl_xor(esum, off, 32);
          sel  += __shfl_xor(sel,  off, 32);
        }
        if (ln == 0) atomicAdd(&outAcc[row], sel - __logf(esum));
      }
    }
  }
  __syncthreads();
  if (tid < ROWS) out[rowBase + tid] = outAcc[tid];
}

// =====================================================================
extern "C" void kernel_launch(void* const* d_in, const int* in_sizes, int n_in,
                              void* d_out, int out_size, void* d_ws, size_t ws_size,
                              hipStream_t stream) {
  const int*   x  = (const int*)d_in[0];
  const float* W1 = (const float*)d_in[1];
  const float* b1 = (const float*)d_in[2];
  const float* W2 = (const float*)d_in[3];
  const float* b2 = (const float*)d_in[4];
  const float* W3 = (const float*)d_in[5];
  const float* b3 = (const float*)d_in[6];
  float* out = (float*)d_out;

  __bf16* W2t = (__bf16*)d_ws;                    // 2 MB
  __bf16* W3t = W2t + (size_t)HH * HH;            // 2 MB

  dim3 pgrid(HH * HH / 256, 2);
  made_prep_weights<<<pgrid, 256, 0, stream>>>(W2, W3, W2t, W3t);

  size_t shmem = sizeof(int) * XS_INTS +
                 (size_t)2 * ROWS * LSTRIDE * sizeof(__bf16) +
                 ROWS * sizeof(float);
  (void)hipFuncSetAttribute((const void*)made_fused,
                            hipFuncAttributeMaxDynamicSharedMemorySize, (int)shmem);
  made_fused<<<BB / ROWS, 512, shmem, stream>>>(x, W1, b1, b2, b3, W2t, W3t, out);
}
